// DecoderLayer_19816979104174
// MI455X (gfx1250) — matrix operand order repaired and checked by reading, not compiled, not run on hardware
//
#include <hip/hip_runtime.h>
#include <hip/hip_bf16.h>

// ---------------------------------------------------------------------------
// DecoderLayer on gfx1250 (MI455X): bf16 WMMA pipeline, fp32 accumulate.
// Operands pre-converted to bf16 once; all GEMM B operands N-major so every
// fragment load is contiguous 16B/lane. 32x64 tile per wave, software-
// pipelined K loop: next iteration's 12 b128 loads issued before the current
// iteration's 8 v_wmma, so loadcnt waits overlap the matrix pipe.
// B=8 T=1024 S=1024 D=512 H=8 F=2048
// ---------------------------------------------------------------------------

typedef __attribute__((ext_vector_type(16))) __bf16 v16bf;
typedef __attribute__((ext_vector_type(8)))  __bf16 v8bf;
typedef __attribute__((ext_vector_type(8)))  float  v8f;
typedef long long ll;

#define BB 8
#define TT 1024
#define SS 1024
#define DD 512
#define HH 8
#define FF 2048

__device__ __forceinline__ v16bf concat8(v8bf lo, v8bf hi) {
  return __builtin_shufflevector(lo, hi, 0, 1, 2, 3, 4, 5, 6, 7,
                                 8, 9, 10, 11, 12, 13, 14, 15);
}

// One wave computes a 32x64 tile of C = alpha*(A@B) + bias (opt ReLU).
// A: MxK row-major bf16 (lda). Bt: N-major bf16, B[k][n] = Bt[n*ldb + k].
// Batch z: z0 = z/Z1, z1 = z%Z1 with independent operand strides.
// transC stores C[n*ldc + row] (used to produce pre-transposed V).
template <typename OutT>
__global__ __launch_bounds__(32) void wmma_gemm(
    const __bf16* __restrict__ A, const __bf16* __restrict__ Bt,
    const float* __restrict__ bias, OutT* __restrict__ C,
    int K, int lda, int ldb, int ldc, int Z1,
    ll sA0, ll sA1, ll sB0, ll sB1, ll sC0, ll sC1, ll sBias1,
    float alpha, int relu, int transC)
{
  const int lane = threadIdx.x;
  const int half = lane >> 4;       // wave32 half-select per gfx1250 layouts
  const int l16  = lane & 15;
  const int tn = blockIdx.x * 64;
  const int tm = blockIdx.y * 32;
  const int z  = blockIdx.z;
  const int z0 = z / Z1, z1 = z - z0 * Z1;

  A    += z0 * sA0 + z1 * sA1;
  Bt   += z0 * sB0 + z1 * sB1;
  C    += z0 * sC0 + z1 * sC1;
  if (bias) bias += z1 * sBias1;

  v8f acc[2][4];
#pragma unroll
  for (int i = 0; i < 2; ++i)
#pragma unroll
    for (int j = 0; j < 4; ++j)
      acc[i][j] = (v8f){0.f, 0.f, 0.f, 0.f, 0.f, 0.f, 0.f, 0.f};

  // Per-lane base pointers (bumped by 32 elements per K-step).
  // A fragment: lane holds K = {half*8+0..7, 16+half*8+0..7} of its row.
  // B fragment: lane holds K = {half*8+0..7, 16+half*8+0..7} of its column.
  const __bf16* pa0 = A + (ll)(tm + l16) * lda + half * 8;
  const __bf16* pa1 = A + (ll)(tm + 16 + l16) * lda + half * 8;
  const __bf16* pb0 = Bt + (ll)(tn + 0 * 16 + l16) * ldb + half * 8;
  const __bf16* pb1 = Bt + (ll)(tn + 1 * 16 + l16) * ldb + half * 8;
  const __bf16* pb2 = Bt + (ll)(tn + 2 * 16 + l16) * ldb + half * 8;
  const __bf16* pb3 = Bt + (ll)(tn + 3 * 16 + l16) * ldb + half * 8;

  // ---- prologue: load K-step 0 fragments ----
  v16bf a0 = concat8(*(const v8bf*)(pa0), *(const v8bf*)(pa0 + 16));
  v16bf a1 = concat8(*(const v8bf*)(pa1), *(const v8bf*)(pa1 + 16));
  v16bf b0 = concat8(*(const v8bf*)(pb0), *(const v8bf*)(pb0 + 16));
  v16bf b1 = concat8(*(const v8bf*)(pb1), *(const v8bf*)(pb1 + 16));
  v16bf b2 = concat8(*(const v8bf*)(pb2), *(const v8bf*)(pb2 + 16));
  v16bf b3 = concat8(*(const v8bf*)(pb3), *(const v8bf*)(pb3 + 16));

  for (int k0 = 32; k0 < K; k0 += 32) {
    // Issue next iteration's loads first so they are in flight while the
    // matrix pipe consumes the current fragments.
    v8bf na0l = *(const v8bf*)(pa0 + k0);
    v8bf na0h = *(const v8bf*)(pa0 + k0 + 16);
    v8bf na1l = *(const v8bf*)(pa1 + k0);
    v8bf na1h = *(const v8bf*)(pa1 + k0 + 16);
    v8bf nb0l = *(const v8bf*)(pb0 + k0);
    v8bf nb0h = *(const v8bf*)(pb0 + k0 + 16);
    v8bf nb1l = *(const v8bf*)(pb1 + k0);
    v8bf nb1h = *(const v8bf*)(pb1 + k0 + 16);
    v8bf nb2l = *(const v8bf*)(pb2 + k0);
    v8bf nb2h = *(const v8bf*)(pb2 + k0 + 16);
    v8bf nb3l = *(const v8bf*)(pb3 + k0);
    v8bf nb3h = *(const v8bf*)(pb3 + k0 + 16);

    acc[0][0] = __builtin_amdgcn_wmma_f32_16x16x32_bf16(false, a0, false, b0, (short)0, acc[0][0], false, false);
    acc[1][0] = __builtin_amdgcn_wmma_f32_16x16x32_bf16(false, a1, false, b0, (short)0, acc[1][0], false, false);
    acc[0][1] = __builtin_amdgcn_wmma_f32_16x16x32_bf16(false, a0, false, b1, (short)0, acc[0][1], false, false);
    acc[1][1] = __builtin_amdgcn_wmma_f32_16x16x32_bf16(false, a1, false, b1, (short)0, acc[1][1], false, false);
    acc[0][2] = __builtin_amdgcn_wmma_f32_16x16x32_bf16(false, a0, false, b2, (short)0, acc[0][2], false, false);
    acc[1][2] = __builtin_amdgcn_wmma_f32_16x16x32_bf16(false, a1, false, b2, (short)0, acc[1][2], false, false);
    acc[0][3] = __builtin_amdgcn_wmma_f32_16x16x32_bf16(false, a0, false, b3, (short)0, acc[0][3], false, false);
    acc[1][3] = __builtin_amdgcn_wmma_f32_16x16x32_bf16(false, a1, false, b3, (short)0, acc[1][3], false, false);

    a0 = concat8(na0l, na0h);
    a1 = concat8(na1l, na1h);
    b0 = concat8(nb0l, nb0h);
    b1 = concat8(nb1l, nb1h);
    b2 = concat8(nb2l, nb2h);
    b3 = concat8(nb3l, nb3h);
  }
  // ---- epilogue: final K-step ----
  acc[0][0] = __builtin_amdgcn_wmma_f32_16x16x32_bf16(false, a0, false, b0, (short)0, acc[0][0], false, false);
  acc[1][0] = __builtin_amdgcn_wmma_f32_16x16x32_bf16(false, a1, false, b0, (short)0, acc[1][0], false, false);
  acc[0][1] = __builtin_amdgcn_wmma_f32_16x16x32_bf16(false, a0, false, b1, (short)0, acc[0][1], false, false);
  acc[1][1] = __builtin_amdgcn_wmma_f32_16x16x32_bf16(false, a1, false, b1, (short)0, acc[1][1], false, false);
  acc[0][2] = __builtin_amdgcn_wmma_f32_16x16x32_bf16(false, a0, false, b2, (short)0, acc[0][2], false, false);
  acc[1][2] = __builtin_amdgcn_wmma_f32_16x16x32_bf16(false, a1, false, b2, (short)0, acc[1][2], false, false);
  acc[0][3] = __builtin_amdgcn_wmma_f32_16x16x32_bf16(false, a0, false, b3, (short)0, acc[0][3], false, false);
  acc[1][3] = __builtin_amdgcn_wmma_f32_16x16x32_bf16(false, a1, false, b3, (short)0, acc[1][3], false, false);

#pragma unroll
  for (int i = 0; i < 2; ++i) {
#pragma unroll
    for (int j = 0; j < 4; ++j) {
      const int n = tn + j * 16 + l16;
      float bv = 0.f;
      if (bias) bv = bias[n];
#pragma unroll
      for (int r = 0; r < 8; ++r) {
        const int row = tm + i * 16 + half * 8 + r;  // C: VGPR r -> M=half*8+r
        float vv = alpha * acc[i][j][r] + bv;
        if (relu) vv = fmaxf(vv, 0.f);
        if (transC) C[(ll)n * ldc + row] = (OutT)vv;
        else        C[(ll)row * ldc + n] = (OutT)vv;
      }
    }
  }
}

// Elementwise fp32 -> bf16.
__global__ __launch_bounds__(256) void cvt_bf16(
    const float* __restrict__ in, __bf16* __restrict__ out, ll n)
{
  const ll i = (ll)blockIdx.x * 256 + threadIdx.x;
  if (i < n) out[i] = (__bf16)in[i];
}

// Batched fp32 (R,C) -> bf16 (C,R) transpose via LDS tile.
__global__ __launch_bounds__(256) void cvt_bf16_tr(
    const float* __restrict__ in, __bf16* __restrict__ out, int R, int C)
{
  __shared__ __bf16 tile[16][17];
  const ll zoff = (ll)blockIdx.z * R * C;
  in += zoff; out += zoff;
  const int r0 = blockIdx.y * 16, c0 = blockIdx.x * 16;
  const int tx = threadIdx.x & 15, ty = threadIdx.x >> 4;
  tile[ty][tx] = (__bf16)in[(ll)(r0 + ty) * C + c0 + tx];
  __syncthreads();
  out[(ll)(c0 + ty) * R + r0 + tx] = tile[tx][ty];
}

// Row softmax over S with optional per-(b,s) pad mask and per-(t,s) causal
// mask; normalized probabilities written as bf16 for the AV GEMM.
__global__ __launch_bounds__(256) void softmax_mask_rows(
    float* __restrict__ scores, __bf16* __restrict__ probs,
    const float* __restrict__ pad, const float* __restrict__ causal,
    int S, int T, int H)
{
  const int row = blockIdx.x;                 // (b*H + h)*T + t
  const int t  = row % T;
  const int bh = row / T;
  const int b  = bh / H;
  float* r = scores + (ll)row * S;
  __bf16* p = probs + (ll)row * S;
  const float* padrow = pad    ? pad    + (ll)b * S : nullptr;
  const float* crow   = causal ? causal + (ll)t * S : nullptr;

  __shared__ float red[256];
  const int tid = threadIdx.x;

  float mx = -3.0e38f;
  for (int i = tid; i < S; i += 256) {
    float v = r[i];
    if (padrow) v += padrow[i];
    if (crow)   v += crow[i];
    r[i] = v;
    mx = fmaxf(mx, v);
  }
  red[tid] = mx; __syncthreads();
  for (int s = 128; s > 0; s >>= 1) {
    if (tid < s) red[tid] = fmaxf(red[tid], red[tid + s]);
    __syncthreads();
  }
  mx = red[0]; __syncthreads();

  float sum = 0.f;
  for (int i = tid; i < S; i += 256) {
    float e = __expf(r[i] - mx);
    r[i] = e;
    sum += e;
  }
  red[tid] = sum; __syncthreads();
  for (int s = 128; s > 0; s >>= 1) {
    if (tid < s) red[tid] += red[tid + s];
    __syncthreads();
  }
  const float inv = 1.0f / red[0];
  for (int i = tid; i < S; i += 256) p[i] = (__bf16)(r[i] * inv);
}

// out = LayerNorm(a + res)*g + b ; also emits bf16 copy when outb != null.
__global__ __launch_bounds__(256) void residual_layernorm(
    const float* __restrict__ a, const float* __restrict__ res,
    const float* __restrict__ g, const float* __restrict__ bta,
    float* __restrict__ out, __bf16* __restrict__ outb, int D)
{
  const ll row = blockIdx.x;
  const float* ar = a   + row * D;
  const float* rr = res + row * D;
  float* orow     = out + row * D;
  __bf16* obrow   = outb ? outb + row * D : nullptr;

  __shared__ float red[256];
  __shared__ float red2[256];
  const int tid = threadIdx.x;

  float s = 0.f, s2 = 0.f;
  for (int i = tid; i < D; i += 256) {
    const float v = ar[i] + rr[i];
    s += v; s2 += v * v;
  }
  red[tid] = s; red2[tid] = s2; __syncthreads();
  for (int st = 128; st > 0; st >>= 1) {
    if (tid < st) { red[tid] += red[tid + st]; red2[tid] += red2[tid + st]; }
    __syncthreads();
  }
  const float mean = red[0] / D;
  const float var  = red2[0] / D - mean * mean;
  const float inv  = rsqrtf(var + 1e-5f);
  for (int i = tid; i < D; i += 256) {
    const float v = ar[i] + rr[i];
    const float o = (v - mean) * inv * g[i] + bta[i];
    orow[i] = o;
    if (obrow) obrow[i] = (__bf16)o;
  }
}

extern "C" void kernel_launch(void* const* d_in, const int* in_sizes, int n_in,
                              void* d_out, int out_size, void* d_ws, size_t ws_size,
                              hipStream_t stream) {
  (void)in_sizes; (void)n_in; (void)out_size; (void)ws_size;

  const float* x       = (const float*)d_in[0];
  const float* src_pad = (const float*)d_in[1];
  const float* tgt_pad = (const float*)d_in[3];
  const float* causal  = (const float*)d_in[4];
  const float* mem_k   = (const float*)d_in[5];
  const float* mem_v   = (const float*)d_in[6];
  const float* Wq_s    = (const float*)d_in[7];
  const float* bq_s    = (const float*)d_in[8];
  const float* Wk_s    = (const float*)d_in[9];
  const float* bk_s    = (const float*)d_in[10];
  const float* Wv_s    = (const float*)d_in[11];
  const float* bv_s    = (const float*)d_in[12];
  const float* Wo_s    = (const float*)d_in[13];
  const float* bo_s    = (const float*)d_in[14];
  const float* Wq_m    = (const float*)d_in[15];
  const float* bq_m    = (const float*)d_in[16];
  const float* Wo_m    = (const float*)d_in[17];
  const float* bo_m    = (const float*)d_in[18];
  const float* W1      = (const float*)d_in[19];
  const float* b1      = (const float*)d_in[20];
  const float* W2      = (const float*)d_in[21];
  const float* b2      = (const float*)d_in[22];
  const float* ln1_g   = (const float*)d_in[23];
  const float* ln1_b   = (const float*)d_in[24];
  const float* ln2_g   = (const float*)d_in[25];
  const float* ln2_b   = (const float*)d_in[26];
  const float* ln3_g   = (const float*)d_in[27];
  const float* ln3_b   = (const float*)d_in[28];
  float* out = (float*)d_out;

  // ---------------- workspace carving ----------------
  char* wp = (char*)d_ws;
  auto alloc = [&](size_t bytes) -> void* {
    void* r = (void*)wp;
    wp += (bytes + 255) & ~(size_t)255;
    return r;
  };
  const ll nBHTD = (ll)BB * HH * TT * DD;
  const ll nBHTS = (ll)BB * HH * TT * SS;
  const ll nBTHD = (ll)BB * TT * HH * DD;
  const ll nBTD  = (ll)BB * TT * DD;
  const ll nBTF  = (ll)BB * TT * FF;
  const ll nBSD  = (ll)BB * SS * DD;
  const ll nHDD  = (ll)HH * DD * DD;

  float*  sc     = (float*) alloc(nBHTS * 4);     // fp32 attention scores
  float*  tmp    = (float*) alloc(nBTD  * 4);     // pre-LN GEMM outputs
  float*  x1     = (float*) alloc(nBTD  * 4);
  float*  x2     = (float*) alloc(nBTD  * 4);
  __bf16* xb     = (__bf16*)alloc(nBTD  * 2);
  __bf16* qb     = (__bf16*)alloc(nBHTD * 2);     // (B,H,T,D)
  __bf16* kb     = (__bf16*)alloc(nBHTD * 2);     // (B,H,S,D)
  __bf16* vTb    = (__bf16*)alloc(nBHTD * 2);     // (B,H,D,S) pre-transposed
  __bf16* scb    = (__bf16*)alloc(nBHTS * 2);     // bf16 softmax probs
  __bf16* catb   = (__bf16*)alloc(nBTHD * 2);     // (B,T,H*D)
  __bf16* x1b    = (__bf16*)alloc(nBTD  * 2);
  __bf16* x2b    = (__bf16*)alloc(nBTD  * 2);
  __bf16* ffhb   = (__bf16*)alloc(nBTF  * 2);
  __bf16* Wq_st  = (__bf16*)alloc(nHDD * 2);      // per-head (D,D) -> (E,D)
  __bf16* Wk_st  = (__bf16*)alloc(nHDD * 2);
  __bf16* Wv_st  = (__bf16*)alloc(nHDD * 2);
  __bf16* Wo_st  = (__bf16*)alloc((ll)HH * DD * DD * 2);  // (4096,512)->(512,4096)
  __bf16* Wq_mt  = (__bf16*)alloc(nHDD * 2);
  __bf16* Wo_mt  = (__bf16*)alloc((ll)HH * DD * DD * 2);
  __bf16* W1t    = (__bf16*)alloc((ll)DD * FF * 2);       // (512,2048)->(2048,512)
  __bf16* W2t    = (__bf16*)alloc((ll)FF * DD * 2);       // (2048,512)->(512,2048)
  __bf16* mem_kb = (__bf16*)alloc(nBSD * 2);              // (B,S,D) as-is
  __bf16* mem_vT = (__bf16*)alloc(nBSD * 2);              // (B,D,S)

  const float scale = 0.04419417382415922f;  // 1/sqrt(512)
  const dim3 wave(32, 1, 1);

  // ---------------- one-time conversions ----------------
  cvt_bf16<<<(int)((nBTD + 255) / 256), 256, 0, stream>>>(x, xb, nBTD);
  cvt_bf16<<<(int)((nBSD + 255) / 256), 256, 0, stream>>>(mem_k, mem_kb, nBSD);
  cvt_bf16_tr<<<dim3(DD / 16, SS / 16, BB), 256, 0, stream>>>(mem_v, mem_vT, SS, DD);
  cvt_bf16_tr<<<dim3(DD / 16, DD / 16, HH), 256, 0, stream>>>(Wq_s, Wq_st, DD, DD);
  cvt_bf16_tr<<<dim3(DD / 16, DD / 16, HH), 256, 0, stream>>>(Wk_s, Wk_st, DD, DD);
  cvt_bf16_tr<<<dim3(DD / 16, DD / 16, HH), 256, 0, stream>>>(Wv_s, Wv_st, DD, DD);
  cvt_bf16_tr<<<dim3(DD / 16, (HH * DD) / 16, 1), 256, 0, stream>>>(Wo_s, Wo_st, HH * DD, DD);
  cvt_bf16_tr<<<dim3(DD / 16, DD / 16, HH), 256, 0, stream>>>(Wq_m, Wq_mt, DD, DD);
  cvt_bf16_tr<<<dim3(DD / 16, (HH * DD) / 16, 1), 256, 0, stream>>>(Wo_m, Wo_mt, HH * DD, DD);
  cvt_bf16_tr<<<dim3(FF / 16, DD / 16, 1), 256, 0, stream>>>(W1, W1t, DD, FF);
  cvt_bf16_tr<<<dim3(DD / 16, FF / 16, 1), 256, 0, stream>>>(W2, W2t, FF, DD);

  // ===================== self attention =====================
  {  // Q/K projections -> (B,H,T,D); V projection -> transposed (B,H,D,S)
    dim3 g(DD / 64, TT / 32, BB * HH);
    wmma_gemm<__bf16><<<g, wave, 0, stream>>>(xb, Wq_st, bq_s, qb,
        DD, DD, DD, DD, HH,
        (ll)TT * DD, 0, 0, (ll)DD * DD, (ll)HH * TT * DD, (ll)TT * DD, DD,
        1.0f, 0, 0);
    wmma_gemm<__bf16><<<g, wave, 0, stream>>>(xb, Wk_st, bk_s, kb,
        DD, DD, DD, DD, HH,
        (ll)TT * DD, 0, 0, (ll)DD * DD, (ll)HH * TT * DD, (ll)TT * DD, DD,
        1.0f, 0, 0);
    wmma_gemm<__bf16><<<g, wave, 0, stream>>>(xb, Wv_st, bv_s, vTb,
        DD, DD, DD, SS, HH,
        (ll)TT * DD, 0, 0, (ll)DD * DD, (ll)HH * DD * SS, (ll)DD * SS, DD,
        1.0f, 0, /*transC=*/1);
  }
  {  // scores = scale * q @ k^T -> fp32 (B,H,T,S); kb is already N-major
    dim3 g(SS / 64, TT / 32, BB * HH);
    wmma_gemm<float><<<g, wave, 0, stream>>>(qb, kb, nullptr, sc,
        DD, DD, DD, SS, HH,
        (ll)HH * TT * DD, (ll)TT * DD, (ll)HH * SS * DD, (ll)SS * DD,
        (ll)HH * TT * SS, (ll)TT * SS, 0,
        scale, 0, 0);
  }
  softmax_mask_rows<<<BB * HH * TT, 256, 0, stream>>>(sc, scb, tgt_pad, causal, SS, TT, HH);
  {  // cat[b,t,h*D+e] = P @ V  (vTb is N-major by construction)
    dim3 g(DD / 64, TT / 32, BB * HH);
    wmma_gemm<__bf16><<<g, wave, 0, stream>>>(scb, vTb, nullptr, catb,
        SS, SS, SS, HH * DD, HH,
        (ll)HH * TT * SS, (ll)TT * SS, (ll)HH * DD * SS, (ll)DD * SS,
        (ll)TT * HH * DD, DD, 0,
        1.0f, 0, 0);
  }
  {  // sa = cat @ Wo_self + bo_self -> fp32 tmp
    dim3 g(DD / 64, (BB * TT) / 32, 1);
    wmma_gemm<float><<<g, wave, 0, stream>>>(catb, Wo_st, bo_s, tmp,
        HH * DD, HH * DD, HH * DD, DD, 1,
        0, 0, 0, 0, 0, 0, 0,
        1.0f, 0, 0);
  }
  residual_layernorm<<<BB * TT, 256, 0, stream>>>(tmp, x, ln1_g, ln1_b, x1, x1b, DD);

  // ===================== cross attention =====================
  {
    dim3 g(DD / 64, TT / 32, BB * HH);
    wmma_gemm<__bf16><<<g, wave, 0, stream>>>(x1b, Wq_mt, bq_m, qb,
        DD, DD, DD, DD, HH,
        (ll)TT * DD, 0, 0, (ll)DD * DD, (ll)HH * TT * DD, (ll)TT * DD, DD,
        1.0f, 0, 0);
  }
  {
    dim3 g(SS / 64, TT / 32, BB * HH);
    wmma_gemm<float><<<g, wave, 0, stream>>>(qb, mem_kb, nullptr, sc,
        DD, DD, DD, SS, HH,
        (ll)HH * TT * DD, (ll)TT * DD, (ll)SS * DD, 0,
        (ll)HH * TT * SS, (ll)TT * SS, 0,
        scale, 0, 0);
  }
  softmax_mask_rows<<<BB * HH * TT, 256, 0, stream>>>(sc, scb, src_pad, nullptr, SS, TT, HH);
  {
    dim3 g(DD / 64, TT / 32, BB * HH);
    wmma_gemm<__bf16><<<g, wave, 0, stream>>>(scb, mem_vT, nullptr, catb,
        SS, SS, SS, HH * DD, HH,
        (ll)HH * TT * SS, (ll)TT * SS, (ll)DD * SS, 0,
        (ll)TT * HH * DD, DD, 0,
        1.0f, 0, 0);
  }
  {
    dim3 g(DD / 64, (BB * TT) / 32, 1);
    wmma_gemm<float><<<g, wave, 0, stream>>>(catb, Wo_mt, bo_m, tmp,
        HH * DD, HH * DD, HH * DD, DD, 1,
        0, 0, 0, 0, 0, 0, 0,
        1.0f, 0, 0);
  }
  residual_layernorm<<<BB * TT, 256, 0, stream>>>(tmp, x1, ln2_g, ln2_b, x2, x2b, DD);

  // ===================== FFN =====================
  {
    dim3 g(FF / 64, (BB * TT) / 32, 1);
    wmma_gemm<__bf16><<<g, wave, 0, stream>>>(x2b, W1t, b1, ffhb,
        DD, DD, DD, FF, 1,
        0, 0, 0, 0, 0, 0, 0,
        1.0f, /*relu=*/1, 0);
  }
  {
    dim3 g(DD / 64, (BB * TT) / 32, 1);
    wmma_gemm<float><<<g, wave, 0, stream>>>(ffhb, W2t, b2, tmp,
        FF, FF, FF, DD, 1,
        0, 0, 0, 0, 0, 0, 0,
        1.0f, 0, 0);
  }
  residual_layernorm<<<BB * TT, 256, 0, stream>>>(tmp, x2, ln3_g, ln3_b, out, nullptr, DD);
}
